// OneBitLinear_39891656245355
// MI455X (gfx1250) — compile-verified, matrix operand order checked by reading, not run on hardware
//
#include <hip/hip_runtime.h>

// OneBit linear: out = ((x*g) @ S^T) * h + bias
//   x [8192,4096] f32, S [16384,4096] f32 (+/-1), g[4096], h[16384], bias[16384]
// Compute-bound GEMM: 1.10 TFLOP vs 0.94 GB min traffic (AI ~1170 FLOP/B).
// S is exact in bf16 -> v_wmma_f32_16x16x32_bf16 with f32 accumulation.
// Block tile 128x256, 8 wave32s each owning a 64x64 patch => 16 WMMA per
// wave per K-step, so conversion/load VALU hides under the XDL pipe.

#define TOKENS 8192
#define IN_F   4096
#define OUT_F  16384

#define BLK_M 128
#define BLK_N 256
#define BLK_K 32
#define LDS_STRIDE 40   // 32 + 8 pad bf16 elems -> 80B row stride; 16 consecutive
                        // rows hit 16 distinct bank groups for 16B fragment loads

typedef __bf16 v16bf __attribute__((ext_vector_type(16)));
typedef __bf16 v8bf  __attribute__((ext_vector_type(8)));
typedef __bf16 v4bf  __attribute__((ext_vector_type(4)));
typedef float  v8f   __attribute__((ext_vector_type(8)));
typedef float  v4f   __attribute__((ext_vector_type(4)));

__global__ __launch_bounds__(256)
void OneBitLinear_wmma_bf16_kernel(const float* __restrict__ x,
                                   const float* __restrict__ S,
                                   const float* __restrict__ g,
                                   const float* __restrict__ h,
                                   const float* __restrict__ bias,
                                   float* __restrict__ out)
{
    __shared__ __bf16 As[BLK_M * LDS_STRIDE];   // (x*g) tile, [m][k]
    __shared__ __bf16 Bs[BLK_N * LDS_STRIDE];   // S tile, [n][k] (S^T cols contiguous)

    const int tid  = threadIdx.x;
    const int lane = tid & 31;
    const int wave = tid >> 5;           // 8 waves: 2 (M) x 4 (N)
    const int wm   = wave & 1;           // 64-row half of M
    const int wn   = wave >> 1;          // 64-col quarter of N (0..3)

    const int lr = lane & 15;            // row/col inside a 16x16 tile
    const int kh = (lane >> 4) << 4;     // K-half: 0 or 16 (contiguous k order, same for A and B)

    const int tRow0 = blockIdx.y * BLK_M;   // token rows
    const int nCol0 = blockIdx.x * BLK_N;   // output features

    // Cooperative load: 256 threads. A: 128x32 f32 (4 rows x 4 floats / thread),
    // B: 256x32 f32 (8 rows x 4 floats / thread). Wave instruction covers
    // 4 rows x 128B contiguous.
    const int ldRow = tid >> 3;          // 0..31
    const int ldCol = (tid & 7) << 2;    // 0..28 step 4

    v8f acc[4][4] = {};                  // 4 M-tiles x 4 N-tiles of 16x16 f32

    for (int kk = 0; kk < IN_F; kk += BLK_K) {
        // g slice for this thread's 4 columns (shared by all its rows)
        v4f gv = *(const v4f*)(g + kk + ldCol);

        #pragma unroll
        for (int i = 0; i < 4; ++i) {
            const int r = ldRow + i * 32;
            v4f xa = *(const v4f*)(x + (size_t)(tRow0 + r) * IN_F + kk + ldCol);
            v4bf ab;
            #pragma unroll
            for (int j = 0; j < 4; ++j) ab[j] = (__bf16)(xa[j] * gv[j]);  // fuse g
            *(v4bf*)(&As[r * LDS_STRIDE + ldCol]) = ab;
        }
        #pragma unroll
        for (int i = 0; i < 8; ++i) {
            const int r = ldRow + i * 32;
            v4f sa = *(const v4f*)(S + (size_t)(nCol0 + r) * IN_F + kk + ldCol);
            v4bf bb;
            #pragma unroll
            for (int j = 0; j < 4; ++j) bb[j] = (__bf16)(sa[j]);          // +/-1 exact
            *(v4bf*)(&Bs[r * LDS_STRIDE + ldCol]) = bb;
        }

        // Prefetch next K-slice (global_prefetch_b8) while this one is consumed
        if (kk + BLK_K < IN_F) {
            __builtin_prefetch(x + (size_t)(tRow0 + ldRow) * IN_F + kk + BLK_K + ldCol, 0, 1);
            __builtin_prefetch(S + (size_t)(nCol0 + ldRow) * IN_F + kk + BLK_K + ldCol, 0, 1);
            __builtin_prefetch(S + (size_t)(nCol0 + ldRow + 128) * IN_F + kk + BLK_K + ldCol, 0, 1);
        }

        __syncthreads();

        // Fragments in contiguous k order (same for A and B):
        //  A: lanes 0-15 -> M=lr,K=0..15 ; lanes 16-31 -> M=lr,K=16..31
        //  B: lanes 0-15 -> N=lr,K=0..15 ; lanes 16-31 -> N=lr,K=16..31
        v16bf a[4], b[4];
        #pragma unroll
        for (int mi = 0; mi < 4; ++mi) {
            const __bf16* p = &As[(wm * 64 + mi * 16 + lr) * LDS_STRIDE + kh];
            v8bf lo = *(const v8bf*)p;
            v8bf hi = *(const v8bf*)(p + 8);
            a[mi] = __builtin_shufflevector(lo, hi, 0,1,2,3,4,5,6,7,8,9,10,11,12,13,14,15);
        }
        #pragma unroll
        for (int ni = 0; ni < 4; ++ni) {
            const __bf16* p = &Bs[(wn * 64 + ni * 16 + lr) * LDS_STRIDE + kh];
            v8bf lo = *(const v8bf*)p;
            v8bf hi = *(const v8bf*)(p + 8);
            b[ni] = __builtin_shufflevector(lo, hi, 0,1,2,3,4,5,6,7,8,9,10,11,12,13,14,15);
        }

        #pragma unroll
        for (int mi = 0; mi < 4; ++mi)
            #pragma unroll
            for (int ni = 0; ni < 4; ++ni)
                acc[mi][ni] = __builtin_amdgcn_wmma_f32_16x16x32_bf16(
                    /*neg_a=*/false, a[mi], /*neg_b=*/false, b[ni],
                    /*c_mod=*/(short)0, acc[mi][ni],
                    /*reuse_a=*/false, /*reuse_b=*/false);

        __syncthreads();
    }

    // Epilogue: C/D layout = VGPR v: lanes 0-15 -> M=v, lanes 16-31 -> M=v+8,
    // N = lane%16. Fuse h and bias (per-lane column fixed per ni).
    const int rbase = tRow0 + wm * 64 + ((lane >> 4) << 3);
    #pragma unroll
    for (int ni = 0; ni < 4; ++ni) {
        const int colg = nCol0 + wn * 64 + ni * 16 + lr;
        const float hv = h[colg];
        const float bv = bias[colg];
        #pragma unroll
        for (int mi = 0; mi < 4; ++mi) {
            #pragma unroll
            for (int v = 0; v < 8; ++v) {
                const int rowg = rbase + mi * 16 + v;
                out[(size_t)rowg * OUT_F + colg] = acc[mi][ni][v] * hv + bv;
            }
        }
    }
}

extern "C" void kernel_launch(void* const* d_in, const int* in_sizes, int n_in,
                              void* d_out, int out_size, void* d_ws, size_t ws_size,
                              hipStream_t stream) {
    (void)in_sizes; (void)n_in; (void)d_ws; (void)ws_size; (void)out_size;
    const float* x    = (const float*)d_in[0];
    const float* S    = (const float*)d_in[1];
    const float* g    = (const float*)d_in[2];
    const float* h    = (const float*)d_in[3];
    const float* bias = (const float*)d_in[4];
    float* out = (float*)d_out;

    dim3 grid(OUT_F / BLK_N, TOKENS / BLK_M);   // 64 x 64 blocks
    dim3 block(256);
    OneBitLinear_wmma_bf16_kernel<<<grid, block, 0, stream>>>(x, S, g, h, bias, out);
}
